// LSRecurrentScanDrivenBlock_51127290691990
// MI455X (gfx1250) — matrix operand order repaired and checked by reading, not compiled, not run on hardware
//
#include <hip/hip_runtime.h>
#include <hip/hip_bf16.h>
#include <cstdint>

#define BATCH 4
#define SEQT  4096
#define DIM   1024
#define NBLK  16
#define BIP   192      // w_post block input width (3072/16)
#define RANK  64
#define NMEM  128
#define LC    64       // scan chunk length
#define NC    64       // number of chunks (T / LC)
#define NTOK  (BATCH*SEQT)
#define NTILE (NTOK/16)
#define RMS_EPS 1.1920929e-07f

typedef __attribute__((ext_vector_type(16))) _Float16 v16h;
typedef __attribute__((ext_vector_type(8)))  _Float16 h8;
typedef __attribute__((ext_vector_type(4)))  _Float16 h4;
typedef __attribute__((ext_vector_type(8)))  float    v8f;

// fast silu: v * v_rcp_f32(1 + exp(-v))  (avoids IEEE div expansion)
__device__ __forceinline__ float silu_f(float v){
  return v * __builtin_amdgcn_rcpf(1.0f + __expf(-v));
}

__device__ __forceinline__ v8f zero8(){
  v8f z;
  #pragma unroll
  for (int i=0;i<8;i++) z[i]=0.f;
  return z;
}

__device__ __forceinline__ v8f wmma_f16(v16h a, v16h b, v8f c){
  // D = A(16x32 f16) x B(32x16 f16) + C(16x16 f32)
  return __builtin_amdgcn_wmma_f32_16x16x32_f16(false, a, false, b, (short)0, c, false, false);
}

// ---- A-matrix fragment (16x32, f16). ISA layout: lane<16 -> row M=lane,
// halves 0..7 = K k0..k0+7, halves 8..15 = K k0+16..k0+23; lane>=16 gets +8.
__device__ __forceinline__ v16h load_fragA(const _Float16* p, int stride, int k0){
  int lane = threadIdx.x & 31;
  int r  = lane & 15;
  int kb = k0 + ((lane & 16) ? 8 : 0);
  const _Float16* base = p + r*stride + kb;
  h8 lo = *(const h8*)(base);
  h8 hi = *(const h8*)(base + 16);
  v16h v;
  #pragma unroll
  for (int i=0;i<8;i++){ v[i]=lo[i]; v[i+8]=hi[i]; }
  return v;
}

// ---- B-matrix fragment (32x16, f16). Same K order as A: column N=lane&15,
// halves 0..7 = K k0..k0+7, halves 8..15 = K k0+16..k0+23; lane>=16 gets +8.
// p is row-major [N][K] (weight rows = output channels), stride in halves.
__device__ __forceinline__ v16h load_fragB(const _Float16* p, int stride, int k0){
  int lane = threadIdx.x & 31;
  int n  = lane & 15;
  int kb = k0 + ((lane & 16) ? 8 : 0);
  const _Float16* base = p + n*stride + kb;
  h8 lo = *(const h8*)(base);
  h8 hi = *(const h8*)(base + 16);
  v16h v;
  #pragma unroll
  for (int i=0;i<8;i++){ v[i]=lo[i]; v[i+8]=hi[i]; }
  return v;
}

// ===================== setup kernels =====================
__global__ void k_cvt_f16(const float* __restrict__ s, _Float16* __restrict__ d, int n){
  int i = blockIdx.x*blockDim.x + threadIdx.x;
  if (i < n) d[i] = (_Float16)s[i];
}

__global__ void k_vecinit(const float* __restrict__ lAs, const float* __restrict__ ldts,
                          const float* __restrict__ lAd, const float* __restrict__ ldtd,
                          const int* __restrict__ kptr,
                          float* __restrict__ a_seq, float* __restrict__ a_pow,
                          float* __restrict__ gain){
  int d = blockIdx.x*blockDim.x + threadIdx.x;
  if (d >= DIM) return;
  float a = expf(-expf(lAs[d]) * expf(ldts[d]));
  a = fmaxf(a, 1e-6f);                      // matches jnp.log(jnp.maximum(a,1e-6))
  a_seq[d] = a;
  a_pow[d] = expf((float)LC * logf(a));     // a^LC chunk carry multiplier
  float K = (float)kptr[0];
  float g;
  if (d < NMEM) {
    g = K;                                  // a==1 branch of geom_gain
  } else {
    float af = expf(-expf(lAd[d-NMEM]) * expf(ldtd[d-NMEM]));
    float om = 1.0f - af;
    g = (fabsf(om) < 1e-6f) ? K : (1.0f - powf(af, K)) / fmaxf(om, 1e-8f);
  }
  gain[d] = g;
}

// ===================== phase 1: rmsnorm + b_seq/b_depth GEMMs =====================
__global__ __launch_bounds__(256) void k_proj1(const float* __restrict__ x,
    const _Float16* __restrict__ Wseq, const _Float16* __restrict__ Wdep,
    const float* __restrict__ gain,
    float* __restrict__ drive, _Float16* __restrict__ hdep, _Float16* __restrict__ e_out){
  __shared__ __align__(16) _Float16 e_lds[16*DIM];
  __shared__ float red[256];
  __shared__ float scl[16];
  const int tid = threadIdx.x;
  const int g0  = blockIdx.x * 16;
  const int row = tid >> 4, seg = tid & 15;

  // rms_norm(x) for 16 tokens; each thread handles 64 contiguous channels
  const float4* xr = (const float4*)(x + (size_t)(g0+row)*DIM + seg*64);
  float4 xv[16];
  float ss = 0.f;
  #pragma unroll
  for (int i=0;i<16;i++){
    xv[i] = xr[i];
    ss += xv[i].x*xv[i].x + xv[i].y*xv[i].y + xv[i].z*xv[i].z + xv[i].w*xv[i].w;
  }
  red[tid] = ss;
  __syncthreads();
  if (tid < 16){
    float s = 0.f;
    #pragma unroll
    for (int i=0;i<16;i++) s += red[tid*16+i];
    scl[tid] = __builtin_amdgcn_rsqf(s*(1.0f/DIM) + RMS_EPS);
  }
  __syncthreads();
  {
    float sc = scl[row];
    h4* ep = (h4*)(e_lds + row*DIM + seg*64);
    h4* eg = (h4*)(e_out + (size_t)(g0+row)*DIM + seg*64);
    #pragma unroll
    for (int i=0;i<16;i++){
      float4 v = xv[i];
      h4 hv;
      hv[0]=(_Float16)(v.x*sc); hv[1]=(_Float16)(v.y*sc);
      hv[2]=(_Float16)(v.z*sc); hv[3]=(_Float16)(v.w*sc);
      ep[i] = hv; eg[i] = hv;
    }
  }
  __syncthreads();

  // block-diag GEMMs: each of 8 waves owns 2 of the 16 blocks
  const int wave = tid >> 5, lane = tid & 31;
  const int n_l = lane & 15, mb = (lane & 16) ? 8 : 0;
  #pragma unroll
  for (int bb=0; bb<2; bb++){
    int b = wave*2 + bb;
    const _Float16* ap = e_lds + b*64;
    #pragma unroll
    for (int nt=0; nt<4; nt++){
      v8f as = zero8(), ad = zero8();
      #pragma unroll
      for (int ks=0; ks<2; ks++){
        v16h A  = load_fragA(ap, DIM, ks*32);
        v16h Bs = load_fragB(Wseq + b*4096 + nt*16*64, 64, ks*32);
        v16h Bd = load_fragB(Wdep + b*4096 + nt*16*64, 64, ks*32);
        as = wmma_f16(A, Bs, as);
        ad = wmma_f16(A, Bd, ad);
      }
      int col = b*64 + nt*16 + n_l;
      float gv = gain[col];
      #pragma unroll
      for (int j=0;j<8;j++){
        int m = mb + j;
        size_t off = (size_t)(g0+m)*DIM + col;
        drive[off] = silu_f(as[j]);                    // scan input, fp32
        hdep[off]  = (_Float16)(gv * silu_f(ad[j]));   // h_depth, f16
      }
    }
  }
}

// ===================== phase 2: blocked scan h[t]=a*h[t-1]+d[t] =====================
__global__ __launch_bounds__(256) void k_scan_local(float* __restrict__ drv,
    const float* __restrict__ a_seq, float* __restrict__ S){
  int tid = blockIdx.x*256 + threadIdx.x;       // (b, c, d) with d fastest -> coalesced
  int d  = tid & (DIM-1);
  int t2 = tid >> 10;
  int c  = t2 & (NC-1);
  int b  = t2 >> 6;
  float a = a_seq[d];
  float* p = drv + ((size_t)(b*SEQT + c*LC))*DIM + d;
  float y = 0.f;
  #pragma unroll 4
  for (int j=0;j<LC;j++){
    y = fmaf(a, y, p[(size_t)j*DIM]);
    p[(size_t)j*DIM] = y;                        // in-place local scan
  }
  S[((size_t)(b*NC + c))*DIM + d] = y;           // chunk-final local state
}

__global__ __launch_bounds__(256) void k_scan_carry(const float* __restrict__ S,
    const float* __restrict__ a_pow, float* __restrict__ P){
  int tid = blockIdx.x*256 + threadIdx.x;        // (b, d)
  int d = tid & (DIM-1);
  int b = tid >> 10;
  float ap = a_pow[d];                           // a^LC
  float F = 0.f;
  for (int c=0;c<NC;c++){
    size_t idx = ((size_t)(b*NC + c))*DIM + d;
    P[idx] = F;                                  // exclusive carry into chunk c
    F = fmaf(ap, F, S[idx]);
  }
}

__global__ __launch_bounds__(256) void k_scan_final(float* __restrict__ drv,
    const _Float16* __restrict__ hdep, const float* __restrict__ a_seq,
    const float* __restrict__ P){
  int tid = blockIdx.x*256 + threadIdx.x;
  int d  = tid & (DIM-1);
  int t2 = tid >> 10;
  int c  = t2 & (NC-1);
  int b  = t2 >> 6;
  float a = a_seq[d];
  float carry = P[((size_t)(b*NC + c))*DIM + d];
  size_t base = ((size_t)(b*SEQT + c*LC))*DIM + d;
  float* p = drv + base;
  const _Float16* q = hdep + base;
  #pragma unroll 4
  for (int j=0;j<LC;j++){
    carry *= a;                                  // h[t] = y_local[t] + a^(j+1)*carry0
    p[(size_t)j*DIM] = p[(size_t)j*DIM] + carry + (float)q[(size_t)j*DIM];  // + h_depth
  }
}

// ===================== phase 3: w_post / w_local / low-rank + output =====================
// Virtual concat [rms_norm(h) | e | shifted_e]: 8-half chunks routed per region.
__device__ __forceinline__ h8 cat_chunk(const _Float16* e_lds, const _Float16* hn_lds,
                                        const _Float16* eprev, int r, int cc){
  if (cc < DIM)   return *(const h8*)(hn_lds + r*DIM + cc);
  if (cc < 2*DIM) return *(const h8*)(e_lds + r*DIM + (cc - DIM));
  int c2 = cc - 2*DIM;
  if (r == 0)     return *(const h8*)(eprev + c2);
  return *(const h8*)(e_lds + (r-1)*DIM + c2);
}

__device__ __forceinline__ v16h load_catA(const _Float16* e_lds, const _Float16* hn_lds,
                                          const _Float16* eprev, int colbase, int k0){
  int lane = threadIdx.x & 31;
  int r  = lane & 15;
  int kb = colbase + k0 + ((lane & 16) ? 8 : 0);
  h8 lo = cat_chunk(e_lds, hn_lds, eprev, r, kb);
  h8 hi = cat_chunk(e_lds, hn_lds, eprev, r, kb + 16);
  v16h v;
  #pragma unroll
  for (int i=0;i<8;i++){ v[i]=lo[i]; v[i+8]=hi[i]; }
  return v;
}

__global__ __launch_bounds__(256) void k_out(const float* __restrict__ x,
    const float* __restrict__ h, const _Float16* __restrict__ e_g,
    const _Float16* __restrict__ Wpost, const _Float16* __restrict__ Wloc,
    const _Float16* __restrict__ lrB, const _Float16* __restrict__ lrA,
    float* __restrict__ out){
  __shared__ __align__(16) unsigned char smem[103488];
  _Float16* e_lds  = (_Float16*)(smem);                 // 32 KB
  _Float16* hn_lds = (_Float16*)(smem + 32768);         // 32 KB
  float*    tpart  = (float*)   (smem + 32768);         // overlays hn after phase A
  _Float16* r_lds  = (_Float16*)(smem + 65536);         // 32 KB
  _Float16* eprev  = (_Float16*)(smem + 98304);         // 2 KB
  _Float16* t_lds  = (_Float16*)(smem + 100352);        // 2 KB
  float*    red    = (float*)   (smem + 102400);        // 1 KB
  float*    scl    = (float*)   (smem + 103424);        // 64 B

  const int tid = threadIdx.x;
  const int g0  = blockIdx.x * 16;
  const int row = tid >> 4, seg = tid & 15;
  const bool bstart = (g0 % SEQT) == 0;

  // load h tile (fp32) + partial sum-of-squares
  const float4* hr = (const float4*)(h + (size_t)(g0+row)*DIM + seg*64);
  float4 hv[16];
  float ss = 0.f;
  #pragma unroll
  for (int i=0;i<16;i++){
    hv[i] = hr[i];
    ss += hv[i].x*hv[i].x + hv[i].y*hv[i].y + hv[i].z*hv[i].z + hv[i].w*hv[i].w;
  }
  red[tid] = ss;
  // copy e tile into LDS (independent of the reduction)
  {
    const h8* es = (const h8*)(e_g + (size_t)(g0+row)*DIM + seg*64);
    h8* ed = (h8*)(e_lds + row*DIM + seg*64);
    #pragma unroll
    for (int i=0;i<8;i++) ed[i] = es[i];
  }
  // shifted_e row for token g0-1 (zeros at batch start)
  {
    int c0 = tid*4;                                       // 256 threads * 4 halves
    h4* dst = (h4*)(eprev + c0);
    if (bstart){
      h4 z; z[0]=(_Float16)0.f; z[1]=(_Float16)0.f; z[2]=(_Float16)0.f; z[3]=(_Float16)0.f;
      *dst = z;
    } else {
      *dst = *(const h4*)(e_g + (size_t)(g0-1)*DIM + c0);
    }
  }
  __syncthreads();
  if (tid < 16){
    float s = 0.f;
    #pragma unroll
    for (int i=0;i<16;i++) s += red[tid*16+i];
    scl[tid] = __builtin_amdgcn_rsqf(s*(1.0f/DIM) + RMS_EPS);
  }
  __syncthreads();
  {
    float sc = scl[row];
    h4* hp = (h4*)(hn_lds + row*DIM + seg*64);
    #pragma unroll
    for (int i=0;i<16;i++){
      float4 v = hv[i];
      h4 hh;
      hh[0]=(_Float16)(v.x*sc); hh[1]=(_Float16)(v.y*sc);
      hh[2]=(_Float16)(v.z*sc); hh[3]=(_Float16)(v.w*sc);
      hp[i] = hh;
    }
  }
  __syncthreads();

  const int wave = tid >> 5, lane = tid & 31;
  const int n_l = lane & 15, mb = (lane & 16) ? 8 : 0;

  // Phase A: r = silu(blockdiag(cat, w_post)); 2 blocks per wave, K=192
  #pragma unroll
  for (int bb=0; bb<2; bb++){
    int b = wave*2 + bb;
    #pragma unroll
    for (int nt=0; nt<4; nt++){
      v8f acc = zero8();
      #pragma unroll
      for (int ks=0; ks<6; ks++){
        v16h A  = load_catA(e_lds, hn_lds, eprev, b*BIP, ks*32);
        v16h Bm = load_fragB(Wpost + b*(64*BIP) + nt*16*BIP, BIP, ks*32);
        acc = wmma_f16(A, Bm, acc);
      }
      int col = b*64 + nt*16 + n_l;
      #pragma unroll
      for (int j=0;j<8;j++) r_lds[(mb+j)*DIM + col] = (_Float16)silu_f(acc[j]);
    }
  }
  __syncthreads();   // hn dead from here; tpart may overlay it

  // Phase B: w_local blockdiag (kept in registers) + split-K r@lr_B^T partials
  v8f accL[2][4];
  #pragma unroll
  for (int bb=0; bb<2; bb++){
    int b = wave*2 + bb;
    #pragma unroll
    for (int nt=0; nt<4; nt++){
      v8f acc = zero8();
      #pragma unroll
      for (int ks=0; ks<2; ks++){
        v16h A  = load_fragA(r_lds + b*64, DIM, ks*32);
        v16h Bm = load_fragB(Wloc + b*4096 + nt*16*64, 64, ks*32);
        acc = wmma_f16(A, Bm, acc);
      }
      accL[bb][nt] = acc;
    }
  }
  #pragma unroll
  for (int nt=0; nt<4; nt++){
    v8f acc = zero8();
    #pragma unroll
    for (int ks=0; ks<4; ks++){
      int kg = wave*128 + ks*32;            // split K=1024 across 8 waves
      v16h A = load_fragA(r_lds, DIM, kg);
      v16h B = load_fragB(lrB + nt*16*DIM, DIM, kg);
      acc = wmma_f16(A, B, acc);
    }
    #pragma unroll
    for (int j=0;j<8;j++)
      tpart[wave*1024 + (mb+j)*64 + nt*16 + n_l] = acc[j];
  }
  __syncthreads();

  // Phase C: reduce 8 wave partials -> t (16x64) in f16
  for (int idx = tid; idx < 1024; idx += 256){
    float s = 0.f;
    #pragma unroll
    for (int w=0; w<8; w++) s += tpart[w*1024 + idx];
    t_lds[idx] = (_Float16)s;
  }
  __syncthreads();

  // Phase D: out = x + local + t @ lr_A^T ; wave owns columns [wave*128, wave*128+128)
  #pragma unroll
  for (int nt8=0; nt8<8; nt8++){
    int n0 = wave*128 + nt8*16;
    v8f acc = zero8();
    #pragma unroll
    for (int ks=0; ks<2; ks++){
      v16h A = load_fragA(t_lds, 64, ks*32);
      v16h B = load_fragB(lrA + n0*64, 64, ks*32);
      acc = wmma_f16(A, B, acc);
    }
    v8f aL = accL[nt8>>2][nt8&3];
    int col = n0 + n_l;
    #pragma unroll
    for (int j=0;j<8;j++){
      int m = mb + j;
      size_t off = (size_t)(g0+m)*DIM + col;
      out[off] = x[off] + aL[j] + acc[j];
    }
  }
}

// ===================== host launch =====================
extern "C" void kernel_launch(void* const* d_in, const int* in_sizes, int n_in,
                              void* d_out, int out_size, void* d_ws, size_t ws_size,
                              hipStream_t stream) {
  const float* x    = (const float*)d_in[0];
  const int*   ak   = (const int*)  d_in[1];
  const float* bseq = (const float*)d_in[2];
  const float* bdep = (const float*)d_in[3];
  const float* wpo  = (const float*)d_in[4];
  const float* wlo  = (const float*)d_in[5];
  const float* lrAf = (const float*)d_in[6];
  const float* lrBf = (const float*)d_in[7];
  const float* lAs  = (const float*)d_in[8];
  const float* ldts = (const float*)d_in[9];
  const float* lAd  = (const float*)d_in[10];
  const float* ldtd = (const float*)d_in[11];
  float* out = (float*)d_out;
  char*  ws  = (char*)d_ws;

  float*    drv   = (float*)(ws);                                 // 64 MB: drive -> local scan -> h
  _Float16* hdep  = (_Float16*)(ws + ((size_t)64  << 20));        // 32 MB
  _Float16* e_h   = (_Float16*)(ws + ((size_t)96  << 20));        // 32 MB
  float*    S     = (float*)   (ws + ((size_t)128 << 20));        // 1 MB
  float*    P     = (float*)   (ws + ((size_t)129 << 20));        // 1 MB
  float*    a_seq = (float*)   (ws + ((size_t)130 << 20));
  float*    a_pow = a_seq + DIM;
  float*    gain  = a_pow + DIM;
  _Float16* wseq_h  = (_Float16*)(ws + ((size_t)130 << 20) + 16384);
  _Float16* wdep_h  = wseq_h  + NBLK*64*64;
  _Float16* wpost_h = wdep_h  + NBLK*64*64;
  _Float16* wloc_h  = wpost_h + NBLK*64*BIP;
  _Float16* lrB_h   = wloc_h  + NBLK*64*64;
  _Float16* lrA_h   = lrB_h   + RANK*DIM;

  // one-time (per-call, deterministic) weight conversions + decay/gain vectors
  k_cvt_f16<<<(NBLK*64*64  + 255)/256, 256, 0, stream>>>(bseq, wseq_h,  NBLK*64*64);
  k_cvt_f16<<<(NBLK*64*64  + 255)/256, 256, 0, stream>>>(bdep, wdep_h,  NBLK*64*64);
  k_cvt_f16<<<(NBLK*64*BIP + 255)/256, 256, 0, stream>>>(wpo,  wpost_h, NBLK*64*BIP);
  k_cvt_f16<<<(NBLK*64*64  + 255)/256, 256, 0, stream>>>(wlo,  wloc_h,  NBLK*64*64);
  k_cvt_f16<<<(RANK*DIM    + 255)/256, 256, 0, stream>>>(lrBf, lrB_h,   RANK*DIM);
  k_cvt_f16<<<(DIM*RANK    + 255)/256, 256, 0, stream>>>(lrAf, lrA_h,   DIM*RANK);
  k_vecinit<<<DIM/256, 256, 0, stream>>>(lAs, ldts, lAd, ldtd, ak, a_seq, a_pow, gain);

  k_proj1<<<NTILE, 256, 0, stream>>>(x, wseq_h, wdep_h, gain, drv, hdep, e_h);
  k_scan_local<<<(BATCH*DIM*NC)/256, 256, 0, stream>>>(drv, a_seq, S);
  k_scan_carry<<<(BATCH*DIM)/256, 256, 0, stream>>>(S, a_pow, P);
  k_scan_final<<<(BATCH*DIM*NC)/256, 256, 0, stream>>>(drv, hdep, a_seq, P);
  k_out<<<NTILE, 256, 0, stream>>>(x, drv, e_h, wpost_h, wloc_h, lrB_h, lrA_h, out);
}
